// HMM_80307298501278
// MI455X (gfx1250) — matrix operand order repaired and checked by reading, not compiled, not run on hardware
//
#include <hip/hip_runtime.h>
#include <hip/hip_bf16.h>

// ---------------------------------------------------------------------------
// HMM forward-backward + Viterbi for MI455X (gfx1250, wave32, WMMA).
//   Y=256 states, X=16 symbols, T=2048 steps.
//
// Roofline: xi output = 537MB writes -> ~23us floor @ 23.3TB/s (NT-store
// streamer). The scans are 2048-long latency chains; per step each wave does
// only 8 ds_load_b128 (operand build via loop-invariant selected pointers,
// zero-pad trick instead of mask multiplies), 4 chained
// v_wmma_f32_16x16x32_f16, 1 LDS partial store, 2 barriers.
// A lives in persistent WMMA VGPR fragments (f16) for the whole loop.
// ---------------------------------------------------------------------------

#define Y_SIZE 256
#define X_SIZE 16
#define SEQ_LEN 2048

typedef __attribute__((ext_vector_type(16))) _Float16 v16h;
typedef __attribute__((ext_vector_type(8)))  _Float16 v8h;
typedef __attribute__((ext_vector_type(8)))  float    v8f;
typedef __attribute__((ext_vector_type(4)))  float    f4;

// ---- output layout (floats, concatenated in reference return order) ----
#define ALPHA_OFF 0
#define BETA_OFF  (SEQ_LEN * Y_SIZE)                        // 524288
#define P_OFF     (2 * SEQ_LEN * Y_SIZE)                    // 1048576
#define GAMMA_OFF (P_OFF + 1)                               // 1048577
#define XI_OFF    (GAMMA_OFF + SEQ_LEN * Y_SIZE)            // 1572865
#define PATH_OFF  (XI_OFF + (size_t)(SEQ_LEN - 1) * Y_SIZE * Y_SIZE)

// ---- workspace layout (byte offsets) ----
#define WS_BX    0u                         // T*Y f32          = 2,097,152 B
#define WS_AH    2097152u                   // Y*Y f16          =   131,072 B
#define WS_BP    2228224u                   // (T-1)*Y u16      = 1,048,064 B
#define WS_VLAST 3276544u                   // Y f32
#define WS_LAST  3277568u                   // 1 int
#define WS_U     3277824u                   // T*Y u16          = 1,048,576 B
#define WS_V     4326400u                   // T*Y u16
// total ~5.4 MB

// ===========================================================================
// Prep: bx[t][y] = b[y, x[t]]   and   Ah = (f16)A  (row-major)
// ===========================================================================
__global__ void hmm_prep(const int* __restrict__ x, const float* __restrict__ A,
                         const float* __restrict__ b, float* __restrict__ bx,
                         _Float16* __restrict__ Ah) {
  const int blk = blockIdx.x, tid = threadIdx.x;
  if (blk < SEQ_LEN) {
    bx[blk * Y_SIZE + tid] = b[tid * X_SIZE + x[blk]];
  } else {
    const int r = blk - SEQ_LEN;
    Ah[r * Y_SIZE + tid] = (_Float16)A[r * Y_SIZE + tid];
  }
}

// helper: load 16 f16 as two selected 16B LDS chunks into a v16h
__device__ __forceinline__ v16h load_frag(const _Float16* p0, const _Float16* p1) {
  const v8h lo = *(const v8h*)p0;
  const v8h hi = *(const v8h*)p1;
  v16h r;
#pragma unroll
  for (int e = 0; e < 8; ++e) { r[e] = lo[e]; r[8 + e] = hi[e]; }
  return r;
}

// ===========================================================================
// Scan kernel: 3 blocks x 1024 threads (32 waves).
//   block 0: forward   alpha[t] = (alpha[t-1] @ A) * bx[t]     (WMMA)
//   block 1: backward  beta[t]  = A @ (bx[t+1] * beta[t+1])    (WMMA)
//   block 2: Viterbi   max-product scan (VALU, A slice in VGPRs)
//
// WMMA fragment mapping (f16 16x16x32, per CDNA5 ISA), L=lane>>4:
//   A-op: lane m=lane&15 holds row m; elems = halves [K=L*8..+7] and
//         [K=16+L*8..+7]  (two contiguous 16B chunks).
//   B-op: lane n=lane&15 holds col n; elems = halves [K=L*8..+7], [K=16+L*8..+7]
//         (two contiguous 16B chunks).
//   D:    VGPR r, lanes 0-15 -> M=r, lanes 16-31 -> M=r+8, N=lane&15.
// ===========================================================================
__global__ __launch_bounds__(1024) void hmm_scan(
    const float* __restrict__ A, const float* __restrict__ pi,
    const float* __restrict__ bx, const _Float16* __restrict__ Ah,
    float* __restrict__ alpha, float* __restrict__ beta,
    float* __restrict__ pOut, unsigned short* __restrict__ bp,
    float* __restrict__ vlast) {
  __shared__ __align__(16) _Float16 sH[Y_SIZE];   // f16 vector for WMMA build
  __shared__ __align__(16) _Float16 sZ[16];       // 32B zero pad (inactive lanes)
  __shared__ __align__(16) float    sP[2 * Y_SIZE]; // parity partial sums
  __shared__ float sPB[1024];                     // Viterbi partial best
  __shared__ int   sPI[1024];                     // Viterbi partial argmax

  const int tid = threadIdx.x;
  const int lane = tid & 31;
  const int w = tid >> 5;           // wave id 0..31
  const int n = lane & 15;          // row (A-op) / col (B-op) within tile
  const int L = lane >> 4;          // lane half
  const bool act0 = (n == 0);       // lanes carrying the vector operand

  if (blockIdx.x == 0) {
    // ---------------- forward ----------------
    const int jt = w >> 1;              // output j-tile 0..15
    const int kbase = (w & 1) * 4;      // k-tiles kbase..kbase+3
    // persistent B fragments: A[kt*32 .. +31][jt*16 .. +15] as f16
    v16h bf[4];
#pragma unroll
    for (int q = 0; q < 4; ++q) {
      const int kt = kbase + q;
#pragma unroll
      for (int e = 0; e < 16; ++e)
        bf[q][e] = Ah[(kt * 32 + (e & 7) + ((e & 8) ? 16 : 0) + L * 8) * Y_SIZE + jt * 16 + n];
    }
    // loop-invariant selected pointers for the alpha (A-operand) fragments
    const _Float16* pa[4];
    const _Float16* pb[4];
#pragma unroll
    for (int q = 0; q < 4; ++q) {
      const int kt = kbase + q;
      pa[q] = act0 ? &sH[kt * 32 + L * 8] : sZ;
      pb[q] = act0 ? &sH[kt * 32 + 16 + L * 8] : sZ;
    }
    float* const pslot = &sP[(w & 1) * Y_SIZE + jt * 16 + (lane & 15)];

    if (tid < 16) sZ[tid] = (_Float16)0.0f;
    if (tid < Y_SIZE) {
      const float a0 = bx[tid] * pi[tid];
      alpha[tid] = a0;
      sH[tid] = (_Float16)a0;
    }
    __syncthreads();

    for (int t = 1; t < SEQ_LEN; ++t) {
      const float bxv = (tid < Y_SIZE) ? bx[t * Y_SIZE + tid] : 0.0f;
      v8f d = {};
#pragma unroll
      for (int q = 0; q < 4; ++q) {
        const v16h af = load_frag(pa[q], pb[q]);
        d = __builtin_amdgcn_wmma_f32_16x16x32_f16(false, af, false, bf[q],
                                                   (short)0, d, false, false);
      }
      // row 0 of D lives in VGPR 0 of d for lanes 0..15 (N = lane)
      if (lane < 16) *pslot = d[0];
      __syncthreads();
      if (tid < Y_SIZE) {
        const float a = (sP[tid] + sP[Y_SIZE + tid]) * bxv;
        alpha[t * Y_SIZE + tid] = a;
        sH[tid] = (_Float16)a;
        if (t == SEQ_LEN - 1) sPB[tid] = a;   // stash exact last row for p
      }
      __syncthreads();
    }
    if (tid == 0) {
      float s = 0.0f;
      for (int y = 0; y < Y_SIZE; ++y) s += sPB[y];
      *pOut = s;
    }
  } else if (blockIdx.x == 1) {
    // ---------------- backward ----------------
    const int it = w >> 1;              // output i-tile
    const int kbase = (w & 1) * 4;
    // persistent A fragments: rows it*16..+15, K = kt*32..+31
    v16h af4[4];
#pragma unroll
    for (int q = 0; q < 4; ++q) {
      const int kt = kbase + q;
#pragma unroll
      for (int e = 0; e < 16; ++e) {
        const int K = (e & 7) + ((e & 8) ? 16 : 0) + L * 8;
        af4[q][e] = Ah[(it * 16 + n) * Y_SIZE + kt * 32 + K];
      }
    }
    // loop-invariant selected pointers for the w (B-operand) fragments
    const _Float16* pa[4];
    const _Float16* pb[4];
#pragma unroll
    for (int q = 0; q < 4; ++q) {
      const int kt = kbase + q;
      pa[q] = act0 ? &sH[kt * 32 + L * 8] : sZ;
      pb[q] = act0 ? &sH[kt * 32 + 16 + L * 8] : sZ;
    }
    float* const pslot = &sP[(w & 1) * Y_SIZE + it * 16 + L * 8];

    if (tid < 16) sZ[tid] = (_Float16)0.0f;
    if (tid < Y_SIZE) {
      beta[(SEQ_LEN - 1) * Y_SIZE + tid] = 1.0f;
      sH[tid] = (_Float16)bx[(SEQ_LEN - 1) * Y_SIZE + tid];  // w for step T-2
    }
    __syncthreads();

    for (int t = SEQ_LEN - 2; t >= 0; --t) {
      const float bxv = (tid < Y_SIZE) ? bx[t * Y_SIZE + tid] : 0.0f;
      v8f d = {};
#pragma unroll
      for (int q = 0; q < 4; ++q) {
        const v16h bfv = load_frag(pa[q], pb[q]);
        d = __builtin_amdgcn_wmma_f32_16x16x32_f16(false, af4[q], false, bfv,
                                                   (short)0, d, false, false);
      }
      // column 0 of D: lane 0 holds M=0..7 (VGPRs 0..7), lane 16 holds M=8..15
      if (n == 0) {
        f4 loq = {d[0], d[1], d[2], d[3]};
        f4 hiq = {d[4], d[5], d[6], d[7]};
        *(f4*)pslot = loq;
        *(f4*)(pslot + 4) = hiq;
      }
      __syncthreads();
      if (tid < Y_SIZE) {
        const float bv = sP[tid] + sP[Y_SIZE + tid];
        beta[t * Y_SIZE + tid] = bv;
        sH[tid] = (_Float16)(bxv * bv);     // w for step t-1
      }
      __syncthreads();
    }
  } else {
    // ---------------- Viterbi ----------------
    const int j = tid & 255;          // state column
    const int q = tid >> 8;           // i-quarter 0..3
    float Areg[64];                   // A[q*64 .. +63][j] held in VGPRs
#pragma unroll
    for (int m = 0; m < 64; ++m) Areg[m] = A[(q * 64 + m) * Y_SIZE + j];
    if (tid < Y_SIZE) sPB[tid] = bx[tid] * pi[tid];  // sPB[0..255] = V vector
    __syncthreads();
    for (int t = 1; t < SEQ_LEN; ++t) {
      const float bxv = (tid < Y_SIZE) ? bx[t * Y_SIZE + tid] : 0.0f;
      float best = -1.0f;
      int bi = 0;
#pragma unroll
      for (int m = 0; m < 64; ++m) {
        const float s = sPB[q * 64 + m] * Areg[m];
        if (s > best) { best = s; bi = q * 64 + m; }
      }
      __syncthreads();                 // all reads of V done
      sPB[256 + tid] = best;           // partials live above the V vector
      sPI[tid] = bi;
      __syncthreads();
      if (tid < Y_SIZE) {
        float bb = sPB[256 + tid];
        int ib = sPI[tid];
#pragma unroll
        for (int r = 1; r < 4; ++r) {
          const float c = sPB[256 + r * 256 + tid];
          if (c > bb) { bb = c; ib = sPI[r * 256 + tid]; }
        }
        bp[(t - 1) * Y_SIZE + tid] = (unsigned short)ib;
        sPB[tid] = bxv * bb;           // next V
      }
      __syncthreads();
    }
    if (tid < Y_SIZE) vlast[tid] = sPB[tid];
  }
}

// ===========================================================================
// Viterbi backtrace via pointer-doubling (avoids 2048 dependent loads).
// ===========================================================================
__global__ void hmm_argmax(const float* __restrict__ vlast, int* __restrict__ last) {
  if (threadIdx.x == 0) {
    float b = vlast[0];
    int bi = 0;
    for (int i = 1; i < Y_SIZE; ++i)
      if (vlast[i] > b) { b = vlast[i]; bi = i; }
    *last = bi;
  }
}

__global__ void hmm_dbl_init(const unsigned short* __restrict__ bp,
                             unsigned short* __restrict__ U) {
  const int t = blockIdx.x, s = threadIdx.x;
  U[t * Y_SIZE + s] = (t < SEQ_LEN - 1) ? bp[t * Y_SIZE + s] : (unsigned short)s;
}

__global__ void hmm_dbl_step(const unsigned short* __restrict__ src,
                             unsigned short* __restrict__ dst, int off) {
  const int t = blockIdx.x, s = threadIdx.x;
  int t2 = t + off;
  if (t2 > SEQ_LEN - 1) t2 = SEQ_LEN - 1;
  const unsigned short inner = src[t2 * Y_SIZE + s];
  dst[t * Y_SIZE + s] = src[t * Y_SIZE + inner];
}

__global__ void hmm_path(const unsigned short* __restrict__ H,
                         const int* __restrict__ last, float* __restrict__ path) {
  const int t = blockIdx.x * blockDim.x + threadIdx.x;
  if (t < SEQ_LEN) path[t] = (float)H[t * Y_SIZE + *last];
}

// ===========================================================================
// gamma = alpha*beta/p ;  xi[t,i,j] = alpha[t,i]*A[i,j]*bx[t+1,j]*beta[t+1,j]
// xi is the 537MB streaming write: float4 non-temporal stores, A from L2.
// ===========================================================================
__global__ void hmm_gamma(const float* __restrict__ alpha, const float* __restrict__ beta,
                          const float* __restrict__ pPtr, float* __restrict__ gamma) {
  const int t = blockIdx.x, y = threadIdx.x;
  const float inv = 1.0f / *pPtr;
  gamma[t * Y_SIZE + y] = alpha[t * Y_SIZE + y] * beta[t * Y_SIZE + y] * inv;
}

__global__ void hmm_xi(const float* __restrict__ A, const float* __restrict__ alpha,
                       const float* __restrict__ beta, const float* __restrict__ bx,
                       float* __restrict__ xi) {
  const int t = blockIdx.x, tid = threadIdx.x;
  __shared__ float sa[Y_SIZE], sw[Y_SIZE];
  sa[tid] = alpha[t * Y_SIZE + tid];
  sw[tid] = bx[(t + 1) * Y_SIZE + tid] * beta[(t + 1) * Y_SIZE + tid];
  __syncthreads();
  const int j0 = (tid & 63) * 4;
  const int iq = tid >> 6;
  const f4 w4 = {sw[j0], sw[j0 + 1], sw[j0 + 2], sw[j0 + 3]};
  const size_t base = (size_t)t * Y_SIZE * Y_SIZE;
#pragma unroll 4
  for (int i = iq * 64; i < iq * 64 + 64; ++i) {
    const float a = sa[i];
    const f4 A4 = *(const f4*)(A + i * Y_SIZE + j0);
    f4 r = A4 * a;
    r = r * w4;
    __builtin_nontemporal_store(r, (f4*)(xi + base + (size_t)i * Y_SIZE + j0));
  }
}

// ===========================================================================
extern "C" void kernel_launch(void* const* d_in, const int* in_sizes, int n_in,
                              void* d_out, int out_size, void* d_ws, size_t ws_size,
                              hipStream_t stream) {
  (void)in_sizes; (void)n_in; (void)out_size; (void)ws_size;
  const int*   x  = (const int*)d_in[0];
  const float* A  = (const float*)d_in[1];
  const float* b  = (const float*)d_in[2];
  const float* pi = (const float*)d_in[3];
  float* out = (float*)d_out;
  char*  ws  = (char*)d_ws;

  float*          bx    = (float*)(ws + WS_BX);
  _Float16*       Ah    = (_Float16*)(ws + WS_AH);
  unsigned short* bp    = (unsigned short*)(ws + WS_BP);
  float*          vlast = (float*)(ws + WS_VLAST);
  int*            last  = (int*)(ws + WS_LAST);
  unsigned short* U     = (unsigned short*)(ws + WS_U);
  unsigned short* V     = (unsigned short*)(ws + WS_V);

  hmm_prep<<<SEQ_LEN + Y_SIZE, Y_SIZE, 0, stream>>>(x, A, b, bx, Ah);

  hmm_scan<<<3, 1024, 0, stream>>>(A, pi, bx, Ah,
                                   out + ALPHA_OFF, out + BETA_OFF,
                                   out + P_OFF, bp, vlast);

  hmm_argmax<<<1, 64, 0, stream>>>(vlast, last);
  hmm_dbl_init<<<SEQ_LEN, Y_SIZE, 0, stream>>>(bp, U);
  unsigned short* src = U;
  unsigned short* dst = V;
  for (int k = 0, off = 1; k < 11; ++k, off <<= 1) {
    hmm_dbl_step<<<SEQ_LEN, Y_SIZE, 0, stream>>>(src, dst, off);
    unsigned short* tmp = src; src = dst; dst = tmp;
  }
  hmm_path<<<SEQ_LEN / 256, 256, 0, stream>>>(src, last, out + PATH_OFF);

  hmm_gamma<<<SEQ_LEN, Y_SIZE, 0, stream>>>(out + ALPHA_OFF, out + BETA_OFF,
                                            out + P_OFF, out + GAMMA_OFF);
  hmm_xi<<<SEQ_LEN - 1, Y_SIZE, 0, stream>>>(A, out + ALPHA_OFF, out + BETA_OFF,
                                             bx, out + XI_OFF);
}
